// PairInBatchNegSoftmaxContrastLoss_20151986553520
// MI455X (gfx1250) — matrix operand order repaired and checked by reading, not compiled, not run on hardware
//
#include <hip/hip_runtime.h>

// ---------------------------------------------------------------------------
// PairInBatchNegSoftmaxContrastLoss for MI455X (gfx1250, wave32, WMMA)
//
// loss = mean_i [ logsumexp_j( (a_i . b_j) / T ) - (a_i . b_i)/T ],
// a,b row-normalized (norm clamped at 1e-8), T = 0.05.
//
//   K1: normalize A,B -> bf16 workspace (A pre-scaled by 20*log2e so the
//       GEMM accumulator is directly the exp2 argument) + exact fp32 diag[i]
//   K2: fused bf16 WMMA GEMM (16x16x32) + streaming sum(exp2(acc)),
//       2 A-strips (32 rows) per wave for 2x B-fragment reuse
//   K3: deterministic mean reduction
// ---------------------------------------------------------------------------

typedef __attribute__((ext_vector_type(16))) __bf16 v16bf;
typedef __attribute__((ext_vector_type(8)))  __bf16 v8bf;
typedef __attribute__((ext_vector_type(8)))  float  v8f;

#define NROWS 8192
#define DIM   256
#define INV_T 20.0f
#define LOG2E 1.4426950408889634f

__device__ __forceinline__ unsigned short f32_to_bf16_rne(float x) {
    unsigned u = __float_as_uint(x);
    unsigned r = u + 0x7fffu + ((u >> 16) & 1u);
    return (unsigned short)(r >> 16);
}

// --------------------------- Kernel 1: normalize ---------------------------
// One 256-thread block per row i: normalize A[i,:],B[i,:] in fp32, store bf16
// copies (A scaled by 20*log2e), write exact fp32 diagonal dot product.
__global__ __launch_bounds__(256) void k_normalize(
    const float* __restrict__ A, const float* __restrict__ B,
    unsigned short* __restrict__ a_bf, unsigned short* __restrict__ b_bf,
    float* __restrict__ diag)
{
    const int row  = blockIdx.x;
    const int t    = threadIdx.x;        // == column, DIM == 256
    const int wave = t >> 5;
    const int lane = t & 31;

    float a = A[row * DIM + t];
    float b = B[row * DIM + t];
    float ssa = a * a, ssb = b * b;
    #pragma unroll
    for (int m = 16; m >= 1; m >>= 1) {
        ssa += __shfl_xor(ssa, m, 32);
        ssb += __shfl_xor(ssb, m, 32);
    }
    __shared__ float sA[8], sB[8], sD[8];
    if (lane == 0) { sA[wave] = ssa; sB[wave] = ssb; }
    __syncthreads();
    float tota = 0.f, totb = 0.f;
    #pragma unroll
    for (int w = 0; w < 8; ++w) { tota += sA[w]; totb += sB[w]; }

    const float inva = 1.0f / fmaxf(sqrtf(tota), 1e-8f);
    const float invb = 1.0f / fmaxf(sqrtf(totb), 1e-8f);
    const float an = a * inva;
    const float bn = b * invb;
    // Pre-scale A so WMMA accumulator == exp2 argument: acc = sim*20*log2e.
    a_bf[row * DIM + t] = f32_to_bf16_rne(an * (INV_T * LOG2E));
    b_bf[row * DIM + t] = f32_to_bf16_rne(bn);

    float d = an * bn;                   // exact fp32 diagonal contribution
    #pragma unroll
    for (int m = 16; m >= 1; m >>= 1) d += __shfl_xor(d, m, 32);
    if (lane == 0) sD[wave] = d;
    __syncthreads();
    if (t == 0) {
        float td = 0.f;
        #pragma unroll
        for (int w = 0; w < 8; ++w) td += sD[w];
        diag[row] = td;
    }
}

// ----------------- Kernel 2: fused WMMA GEMM + logsumexp -------------------
// Grid: 32 blocks x 256 threads (8 waves). Wave w owns 32 rows
// R0 = blk*256 + w*32 (two 16-row strips) and sweeps all 8192 columns in
// 16-wide tiles: per tile 8 B fragments feed 16 WMMAs (2x reuse).
__global__ __launch_bounds__(256) void k_gemm_lse(
    const unsigned short* __restrict__ a_bf,
    const unsigned short* __restrict__ b_bf,
    const float* __restrict__ diag,
    float* __restrict__ row_loss)
{
    const int wave = threadIdx.x >> 5;
    const int lane = threadIdx.x & 31;
    const int half = lane >> 4;          // 0: lanes 0-15, 1: lanes 16-31
    const int lmod = lane & 15;
    const int R0   = blockIdx.x * 256 + wave * 32;

    // ---- preload two A strips (ISA 16-bit A 16x32 layout) ----
    // lane l: row = l%16; low half K in {0..7,16..23}, high {8..15,24..31}
    v16bf afrag0[8], afrag1[8];
    {
        const unsigned short* arow0 = a_bf + (unsigned)(R0 + lmod) * DIM;
        const unsigned short* arow1 = arow0 + 16 * DIM;
        #pragma unroll
        for (int ks = 0; ks < 8; ++ks) {
            const int k0 = ks * 32 + half * 8;
            v8bf lo0 = *(const v8bf*)(arow0 + k0);
            v8bf hi0 = *(const v8bf*)(arow0 + k0 + 16);
            v8bf lo1 = *(const v8bf*)(arow1 + k0);
            v8bf hi1 = *(const v8bf*)(arow1 + k0 + 16);
            v16bf f0, f1;
            #pragma unroll
            for (int i = 0; i < 8; ++i) {
                f0[i] = lo0[i]; f0[i + 8] = hi0[i];
                f1[i] = lo1[i]; f1[i + 8] = hi1[i];
            }
            afrag0[ks] = f0;
            afrag1[ks] = f1;
        }
    }

    float s0[8], s1[8];
    #pragma unroll
    for (int k = 0; k < 8; ++k) { s0[k] = 0.f; s1[k] = 0.f; }

    // B fragment (ISA 16-bit B 32x16 layout): lane l -> col l%16,
    // low half K in {0..7,16..23}, high {8..15,24..31} (matches A).
    const unsigned boff = (unsigned)lmod * DIM + (unsigned)(half * 8);

    for (int ct = 0; ct < NROWS / 16; ++ct) {
        const unsigned short* brow = b_bf + (unsigned)ct * (16 * DIM) + boff;
        // hint next column tile into cache (global_prefetch_b8); wrap to
        // avoid a branch (last-iteration prefetch is harmless).
        __builtin_prefetch(
            b_bf + (unsigned)((ct + 1) & (NROWS / 16 - 1)) * (16 * DIM) + boff,
            0, 1);

        // Load all 8 fragments into distinct regs first: one load clause,
        // partial waits overlap with the WMMA burst.
        v16bf bfrag[8];
        #pragma unroll
        for (int ks = 0; ks < 8; ++ks) {
            v8bf blo = *(const v8bf*)(brow + ks * 32);
            v8bf bhi = *(const v8bf*)(brow + ks * 32 + 16);
            #pragma unroll
            for (int i = 0; i < 8; ++i) { bfrag[ks][i] = blo[i]; bfrag[ks][i + 8] = bhi[i]; }
        }

        v8f acc0 = {}, acc1 = {};
        #pragma unroll
        for (int ks = 0; ks < 8; ++ks) {
            acc0 = __builtin_amdgcn_wmma_f32_16x16x32_bf16(
                false, afrag0[ks], false, bfrag[ks], (short)0, acc0, false, false);
            acc1 = __builtin_amdgcn_wmma_f32_16x16x32_bf16(
                false, afrag1[ks], false, bfrag[ks], (short)0, acc1, false, false);
        }

        // acc already == sim * 20 * log2e (folded into A); logits bounded
        // (unit rows) so plain exp2-sum is fp32-safe, no online max needed.
        #pragma unroll
        for (int k = 0; k < 8; ++k) {
            s0[k] += __builtin_amdgcn_exp2f(acc0[k]);
            s1[k] += __builtin_amdgcn_exp2f(acc1[k]);
        }
    }

    // Reduce partial exp-sums across the 16 lanes of each half
    // (xor masks 1..8 never cross bit 4, halves stay independent).
    #pragma unroll
    for (int m = 1; m < 16; m <<= 1) {
        #pragma unroll
        for (int k = 0; k < 8; ++k) {
            s0[k] += __shfl_xor(s0[k], m, 32);
            s1[k] += __shfl_xor(s1[k], m, 32);
        }
    }

    // C layout: VGPR k <-> row 8*half + k within a strip. Lane with
    // lmod == 8*half + k writes rows R0 + lmod (strip 0) and +16 (strip 1).
    const int k = lmod - half * 8;
    if (k >= 0 && k < 8) {
        float sum0 = s0[0], sum1 = s1[0];
        #pragma unroll
        for (int kk = 1; kk < 8; ++kk) {
            sum0 = (k == kk) ? s0[kk] : sum0;
            sum1 = (k == kk) ? s1[kk] : sum1;
        }
        const int r0 = R0 + lmod;
        const int r1 = r0 + 16;
        const float ln2 = 0.6931471805599453f;
        row_loss[r0] = __builtin_amdgcn_logf(sum0) * ln2 - diag[r0] * INV_T;
        row_loss[r1] = __builtin_amdgcn_logf(sum1) * ln2 - diag[r1] * INV_T;
    }
}

// --------------------- Kernel 3: deterministic mean ------------------------
__global__ __launch_bounds__(256) void k_reduce_mean(
    const float* __restrict__ row_loss, float* __restrict__ out)
{
    const int t = threadIdx.x;
    float acc = 0.f;
    for (int i = t; i < NROWS; i += 256) acc += row_loss[i];
    #pragma unroll
    for (int m = 16; m >= 1; m >>= 1) acc += __shfl_xor(acc, m, 32);
    __shared__ float sbuf[8];
    if ((t & 31) == 0) sbuf[t >> 5] = acc;
    __syncthreads();
    if (t == 0) {
        float tot = 0.f;
        #pragma unroll
        for (int w = 0; w < 8; ++w) tot += sbuf[w];
        out[0] = tot * (1.0f / (float)NROWS);
    }
}

// ---------------------------------------------------------------------------
extern "C" void kernel_launch(void* const* d_in, const int* in_sizes, int n_in,
                              void* d_out, int out_size, void* d_ws, size_t ws_size,
                              hipStream_t stream)
{
    (void)in_sizes; (void)n_in; (void)out_size; (void)ws_size;
    const float* A = (const float*)d_in[0];
    const float* B = (const float*)d_in[1];
    float* out = (float*)d_out;

    char* ws = (char*)d_ws;
    unsigned short* a_bf    = (unsigned short*)(ws);                            // 4 MB
    unsigned short* b_bf    = (unsigned short*)(ws + (size_t)4 * 1024 * 1024);  // 4 MB
    float*          diag    = (float*)(ws + (size_t)8 * 1024 * 1024);           // 32 KB
    float*          rowloss = (float*)(ws + (size_t)8 * 1024 * 1024 + 32 * 1024);

    k_normalize<<<NROWS, 256, 0, stream>>>(A, B, a_bf, b_bf, diag);
    k_gemm_lse<<<NROWS / 256, 256, 0, stream>>>(a_bf, b_bf, diag, rowloss);
    k_reduce_mean<<<1, 256, 0, stream>>>(rowloss, out);
}
